// ContraNorm_10797547782384
// MI455X (gfx1250) — matrix operand order repaired and checked by reading, not compiled, not run on hardware
//
#include <hip/hip_runtime.h>
#include <hip/hip_bf16.h>

// ContraNorm fused kernel for MI455X (gfx1250, wave32, WMMA).
// GEMMs run on v_wmma_f32_16x16x32_bf16; softmax needs no max-subtraction
// because sim is a cosine similarity in [-1, 1].
// v2: A-fragments hoisted to registers in pass 1; independent WMMA chains
// (paired column tiles / 4 D-tiles per k-step) to overlap loads with compute.

typedef __attribute__((ext_vector_type(16))) __bf16 v16bf;
typedef __attribute__((ext_vector_type(8)))  __bf16 v8bf;
typedef __attribute__((ext_vector_type(8)))  float  v8f;

#define B_    8
#define N_    2048
#define D_    512
#define SCALE 0.1f
#define EPS   1e-6f
#define PSTR  (N_ + 8)   // padded LDS row stride (bf16 elems): 4112B/row -> conflict-free b128

// ---------------- prep: row norms, bf16 cast, and transpose ----------------
__global__ __launch_bounds__(256) void prep_kernel(const float* __restrict__ x,
                                                   __bf16* __restrict__ xn,
                                                   __bf16* __restrict__ xt) {
  const int lane = threadIdx.x & 31;
  const int wave = threadIdx.x >> 5;
  const int row  = blockIdx.x * 8 + wave;      // global row in [0, B*N)
  const int b    = row >> 11;                  // row / N_
  const int n    = row & (N_ - 1);
  const float* xr = x + (size_t)row * D_;
  float v[16];
  float ss = 0.f;
#pragma unroll
  for (int i = 0; i < 16; ++i) { v[i] = xr[lane + 32 * i]; ss += v[i] * v[i]; }
#pragma unroll
  for (int off = 16; off > 0; off >>= 1) ss += __shfl_xor(ss, off);
  const float inv = rsqrtf(ss);
#pragma unroll
  for (int i = 0; i < 16; ++i) {
    const int d = lane + 32 * i;
    xn[(size_t)row * D_ + d]          = (__bf16)(v[i] * inv);  // normalized, GEMM1 A/B
    xt[((size_t)b * D_ + d) * N_ + n] = (__bf16)v[i];          // x transposed, GEMM2 B
  }
}

// ---------------- main fused kernel: one 16-row tile per workgroup ----------------
__global__ __launch_bounds__(256) void contranorm_kernel(
    const float* __restrict__ x, const float* __restrict__ lnw,
    const float* __restrict__ lnb, const __bf16* __restrict__ xn,
    const __bf16* __restrict__ xt, float* __restrict__ out) {
  extern __shared__ __bf16 P[];     // [16][PSTR] exp(sim) tile (bf16)
  __shared__ float rowsum[16];

  const int lane   = threadIdx.x & 31;
  const int wave   = threadIdx.x >> 5;
  const int lrow   = lane & 15;            // M (A) or N (B/C) index within tile
  const int khalf  = (lane >> 4) * 8;      // C/D-frag: element r holds tile row kh + r
  const int kquart = (lane >> 4) * 16;     // B-frag: lanes 0-15 K=0..15, 16-31 K=16..31

  const int b  = blockIdx.x >> 7;          // blockIdx / (N_/16)
  const int m0 = (blockIdx.x & 127) * 16;  // first row of this tile

  if (threadIdx.x < 16) rowsum[threadIdx.x] = 0.f;
  __syncthreads();

  // ---------- Pass 1: S = Xn_tile * Xn^T, exp, accumulate row sums ----------
  const __bf16* xnb  = xn + (size_t)b * N_ * D_;
  const __bf16* Arow = xnb + (size_t)(m0 + lrow) * D_;

  // A-fragments for the whole 16x512 tile live in registers (16 x 8 VGPRs),
  // loaded once and reused by all 16 column tiles.
  v16bf Afr[16];
#pragma unroll
  for (int kk = 0; kk < 16; ++kk) {
    v8bf alo = *(const v8bf*)(Arow + kk * 32 + kquart);
    v8bf ahi = *(const v8bf*)(Arow + kk * 32 + kquart + 8);
    Afr[kk]  = __builtin_shufflevector(alo, ahi, 0,1,2,3,4,5,6,7,8,9,10,11,12,13,14,15);
  }

  float rs[8];
#pragma unroll
  for (int r = 0; r < 8; ++r) rs[r] = 0.f;

  for (int mb = 0; mb < 16; mb += 2) {     // this wave's 256-column slice, 2 tiles at a time
    const int col0 = wave * 256 + mb * 16;
    const __bf16* Brow0 = xnb + (size_t)(col0 + lrow) * D_;
    const __bf16* Brow1 = Brow0 + (size_t)16 * D_;
    v8f C0 = {}, C1 = {};
#pragma unroll
    for (int kk = 0; kk < 16; ++kk) {
      v16bf b0 = *(const v16bf*)(Brow0 + kk * 32 + kquart);
      v16bf b1 = *(const v16bf*)(Brow1 + kk * 32 + kquart);
      C0 = __builtin_amdgcn_wmma_f32_16x16x32_bf16(false, Afr[kk], false, b0, (short)0, C0, false, false);
      C1 = __builtin_amdgcn_wmma_f32_16x16x32_bf16(false, Afr[kk], false, b1, (short)0, C1, false, false);
    }
#pragma unroll
    for (int r = 0; r < 8; ++r) {
      const float e0 = __expf(C0[r]);      // |sim| <= 1: overflow-safe softmax
      const float e1 = __expf(C1[r]);
      rs[r] += e0 + e1;
      P[(size_t)(r + khalf) * PSTR + (col0 + lrow)]      = (__bf16)e0;
      P[(size_t)(r + khalf) * PSTR + (col0 + 16 + lrow)] = (__bf16)e1;
    }
  }
#pragma unroll
  for (int off = 1; off < 16; off <<= 1) {
#pragma unroll
    for (int r = 0; r < 8; ++r) rs[r] += __shfl_xor(rs[r], off);
  }
  if (lrow == 0) {
#pragma unroll
    for (int r = 0; r < 8; ++r) atomicAdd(&rowsum[r + khalf], rs[r]);
  }
  __syncthreads();

  float rinv[8];
#pragma unroll
  for (int r = 0; r < 8; ++r) rinv[r] = 1.0f / rowsum[r + khalf];

  // ---------- Pass 2: x_neg = P * X (this wave owns 64 of the 512 D-cols) ----------
  // k outer / 4 D-tiles inner: each LDS A-frag feeds 4 independent WMMAs.
  const __bf16* xtb  = xt + (size_t)b * D_ * N_;
  const __bf16* Bc0 = xtb + (size_t)(wave * 64 + lrow) * N_;
  const __bf16* Bc1 = Bc0 + (size_t)16 * N_;
  const __bf16* Bc2 = Bc0 + (size_t)32 * N_;
  const __bf16* Bc3 = Bc0 + (size_t)48 * N_;
  v8f D0 = {}, D1 = {}, D2 = {}, D3 = {};
#pragma unroll 4
  for (int k = 0; k < N_; k += 32) {
    v8bf alo = *(const v8bf*)(&P[(size_t)lrow * PSTR + k + kquart]);
    v8bf ahi = *(const v8bf*)(&P[(size_t)lrow * PSTR + k + kquart + 8]);
    v16bf a  = __builtin_shufflevector(alo, ahi, 0,1,2,3,4,5,6,7,8,9,10,11,12,13,14,15);
    v16bf b0 = *(const v16bf*)(Bc0 + k + kquart);
    v16bf b1 = *(const v16bf*)(Bc1 + k + kquart);
    v16bf b2 = *(const v16bf*)(Bc2 + k + kquart);
    v16bf b3 = *(const v16bf*)(Bc3 + k + kquart);
    D0 = __builtin_amdgcn_wmma_f32_16x16x32_bf16(false, a, false, b0, (short)0, D0, false, false);
    D1 = __builtin_amdgcn_wmma_f32_16x16x32_bf16(false, a, false, b1, (short)0, D1, false, false);
    D2 = __builtin_amdgcn_wmma_f32_16x16x32_bf16(false, a, false, b2, (short)0, D2, false, false);
    D3 = __builtin_amdgcn_wmma_f32_16x16x32_bf16(false, a, false, b3, (short)0, D3, false, false);
  }
  float acc[4][8];
#pragma unroll
  for (int r = 0; r < 8; ++r) {
    acc[0][r] = D0[r]; acc[1][r] = D1[r]; acc[2][r] = D2[r]; acc[3][r] = D3[r];
  }
  __syncthreads();                          // all waves done reading P

  // ---------- Residual: y = x - SCALE * x_neg / rowsum (into LDS, reusing P) ----------
  float* Y = (float*)P;                     // 16 x 512 f32 = 32KB, fits in P region
  const float* xb = x + ((size_t)b * N_ + m0) * D_;
#pragma unroll
  for (int nb = 0; nb < 4; ++nb) {
    const int dcol = wave * 64 + nb * 16 + lrow;
#pragma unroll
    for (int r = 0; r < 8; ++r) {
      const int row = r + khalf;
      const float y = xb[(size_t)row * D_ + dcol] - SCALE * acc[nb][r] * rinv[r];
      Y[row * D_ + dcol] = y;
    }
  }
  __syncthreads();

  // ---------- LayerNorm over D: each wave handles 2 rows ----------
#pragma unroll
  for (int rr = 0; rr < 2; ++rr) {
    const int row = wave * 2 + rr;
    float vbuf[16];
    float s = 0.f, s2 = 0.f;
#pragma unroll
    for (int i = 0; i < 16; ++i) {
      const float v = Y[row * D_ + lane + 32 * i];
      vbuf[i] = v; s += v; s2 += v * v;
    }
#pragma unroll
    for (int off = 16; off > 0; off >>= 1) {
      s  += __shfl_xor(s, off);
      s2 += __shfl_xor(s2, off);
    }
    const float mu   = s * (1.f / D_);
    const float var  = s2 * (1.f / D_) - mu * mu;
    const float rstd = rsqrtf(var + EPS);
    float* orow = out + ((size_t)b * N_ + m0 + row) * D_;
#pragma unroll
    for (int i = 0; i < 16; ++i) {
      const int dd = lane + 32 * i;
      orow[dd] = (vbuf[i] - mu) * rstd * lnw[dd] + lnb[dd];
    }
  }
}

extern "C" void kernel_launch(void* const* d_in, const int* in_sizes, int n_in,
                              void* d_out, int out_size, void* d_ws, size_t ws_size,
                              hipStream_t stream) {
  const float* x   = (const float*)d_in[0];
  const float* lnw = (const float*)d_in[1];
  const float* lnb = (const float*)d_in[2];
  float* out = (float*)d_out;

  __bf16* xn = (__bf16*)d_ws;                         // 16 MB
  __bf16* xt = xn + (size_t)B_ * N_ * D_;             // 16 MB

  prep_kernel<<<B_ * N_ / 8, 256, 0, stream>>>(x, xn, xt);

  const size_t smem = (size_t)16 * PSTR * sizeof(__bf16);  // 65792 B
  (void)hipFuncSetAttribute((const void*)contranorm_kernel,
                            hipFuncAttributeMaxDynamicSharedMemorySize, (int)smem);
  contranorm_kernel<<<B_ * (N_ / 16), 256, smem, stream>>>(x, lnw, lnb, xn, xt, out);
}
